// BiLSTM_CRF_89670327206340
// MI455X (gfx1250) — matrix operand order repaired and checked by reading, not compiled, not run on hardware
//
#include <hip/hip_runtime.h>
#include <hip/hip_bf16.h>

#define T_SEQ 16384
#define E_DIM 512
#define HHD   512
#define G4    2048
#define NT    5
#define START_TAG 3
#define STOP_TAG  4
#define NEGV (-10000.0f)

typedef __attribute__((ext_vector_type(16))) __bf16 v16bf;
typedef __attribute__((ext_vector_type(8)))  __bf16 v8bf;
typedef __attribute__((ext_vector_type(8)))  float  v8f;

// Pointer types matching the async-to-LDS builtin signature:
// (int4 addrspace(1)*, int4 addrspace(3)*, imm, imm)
typedef int v4i_vs __attribute__((vector_size(16)));
typedef __attribute__((address_space(1))) v4i_vs* glob_v4i_p;
typedef __attribute__((address_space(3))) v4i_vs* lds_v4i_p;

// LDS row stride for W_hh slice (bf16 elems). 520 -> 1040 B rows: 16B-aligned
// for b128 async chunks; word stride 260 => lanes map to 16 distinct banks
// (2-way conflict), acceptable for the alignment win.
#define WROW 520

#if defined(__AMDGCN__) && __has_builtin(__builtin_amdgcn_global_load_async_to_lds_b128) && __has_builtin(__builtin_amdgcn_s_wait_asynccnt)
#define HAVE_ASYNC_LDS 1
#else
#define HAVE_ASYNC_LDS 0
#endif

__device__ __forceinline__ float bf_lo(unsigned int p) {
  return __uint_as_float(p << 16);
}
__device__ __forceinline__ float bf_hi(unsigned int p) {
  return __uint_as_float(p & 0xffff0000u);
}

// ---------------------------------------------------------------------------
// K0: init combined biases, h double-buffer, inter-WG barrier counters
// ---------------------------------------------------------------------------
__global__ void init_misc_kernel(const float* __restrict__ bihf, const float* __restrict__ bhhf,
                                 const float* __restrict__ bihb, const float* __restrict__ bhhb,
                                 const float* __restrict__ h0,
                                 float* __restrict__ biasc,   // [2][2048]
                                 float* __restrict__ hbuf,    // [2][2][512]
                                 unsigned int* __restrict__ bar) {
  int g = blockIdx.x * blockDim.x + threadIdx.x;
  if (g < G4) {
    biasc[g] = bihf[g] + bhhf[g];
  } else if (g < 2 * G4) {
    int i = g - G4;
    biasc[G4 + i] = bihb[i] + bhhb[i];
  }
  if (g < 2 * HHD) {
    int dir = g >> 9, k = g & (HHD - 1);
    hbuf[dir * 2 * HHD + k] = h0[g];   // buffer 0 holds h0 per direction
  }
  if (g < 2) bar[g] = 0u;
}

// ---------------------------------------------------------------------------
// K1: embedding gather + f32->bf16
// ---------------------------------------------------------------------------
__global__ void gather_embed_kernel(const int* __restrict__ sent,
                                    const float* __restrict__ embed,
                                    __bf16* __restrict__ xb) {
  int i = blockIdx.x * blockDim.x + threadIdx.x;
  if (i >= T_SEQ * E_DIM) return;
  int t = i >> 9, e = i & (E_DIM - 1);
  xb[i] = (__bf16)embed[(size_t)sent[t] * E_DIM + e];
}

// ---------------------------------------------------------------------------
// K2: all four weight matrices f32 -> bf16, row-major.
//     Layout: [wih_f | wih_b | whh_f | whh_b], each 2048x512.
// ---------------------------------------------------------------------------
__global__ void conv_weights_kernel(const float* __restrict__ wihf,
                                    const float* __restrict__ wihb,
                                    const float* __restrict__ whhf,
                                    const float* __restrict__ whhb,
                                    __bf16* __restrict__ out) {
  int i = blockIdx.x * blockDim.x + threadIdx.x;
  const int N1 = G4 * E_DIM;
  if (i < N1)              out[i] = (__bf16)wihf[i];
  else if (i < 2 * N1)     out[i] = (__bf16)wihb[i - N1];
  else if (i < 3 * N1)     out[i] = (__bf16)whhf[i - 2 * N1];
  else if (i < 4 * N1)     out[i] = (__bf16)whhb[i - 3 * N1];
}

// ---------------------------------------------------------------------------
// K3: XP[dir] = X @ W_ih[dir]^T + bias   via v_wmma_f32_16x16x32_bf16
//     block = 256 thr (8 waves, 2x4), block tile 64(M) x 128(N), K loop 512/32
// ---------------------------------------------------------------------------
__global__ void __launch_bounds__(256)
gemm_xproj_kernel(const __bf16* __restrict__ X,    // [T][512]
                  const __bf16* __restrict__ Wih,  // [2][2048][512]
                  const float*  __restrict__ biasc,// [2][2048]
                  float* __restrict__ XP) {        // [2][T][2048]
  const int dir = blockIdx.z;
  const __bf16* W   = Wih   + (size_t)dir * G4 * E_DIM;
  const float*  bias= biasc + (size_t)dir * G4;
  float*        xp  = XP    + (size_t)dir * T_SEQ * G4;

  const int lane = threadIdx.x & 31;
  const int wav  = threadIdx.x >> 5;
  const int wm = wav & 1, wn = wav >> 1;
  const int M0 = blockIdx.x * 64  + wm * 32;
  const int N0 = blockIdx.y * 128 + wn * 32;

  const int  l15 = lane & 15;
  const bool hi  = lane >= 16;

  v8f c00 = {}, c01 = {}, c10 = {}, c11 = {};

  // A: lanes 0-15 hold row M, K 0..15; lanes 16-31 K 16..31
  const __bf16* arow0 = X + (size_t)(M0 + l15) * E_DIM + (hi ? 16 : 0);
  const __bf16* arow1 = arow0 + (size_t)16 * E_DIM;
  // B: lanes 0-15 hold col N, K 0..15; lanes 16-31 K 16..31 (row-major W)
  const __bf16* brow0 = W + (size_t)(N0 + l15) * E_DIM + (hi ? 16 : 0);
  const __bf16* brow1 = brow0 + (size_t)16 * E_DIM;

  for (int kt = 0; kt < E_DIM; kt += 32) {
    if (kt + 32 < E_DIM) {
      __builtin_prefetch(arow0 + kt + 32, 0, 1);
      __builtin_prefetch(brow0 + kt + 32, 0, 1);
    }
    v8bf lo0 = *(const v8bf*)(arow0 + kt);
    v8bf hh0 = *(const v8bf*)(arow0 + kt + 8);
    v8bf lo1 = *(const v8bf*)(arow1 + kt);
    v8bf hh1 = *(const v8bf*)(arow1 + kt + 8);
    v16bf a0, a1;
#pragma unroll
    for (int i = 0; i < 8; ++i) {
      a0[i] = lo0[i]; a0[8 + i] = hh0[i];
      a1[i] = lo1[i]; a1[8 + i] = hh1[i];
    }
    v16bf b0 = *(const v16bf*)(brow0 + kt);
    v16bf b1 = *(const v16bf*)(brow1 + kt);
    c00 = __builtin_amdgcn_wmma_f32_16x16x32_bf16(false, a0, false, b0, (short)0, c00, false, false);
    c01 = __builtin_amdgcn_wmma_f32_16x16x32_bf16(false, a0, false, b1, (short)0, c01, false, false);
    c10 = __builtin_amdgcn_wmma_f32_16x16x32_bf16(false, a1, false, b0, (short)0, c10, false, false);
    c11 = __builtin_amdgcn_wmma_f32_16x16x32_bf16(false, a1, false, b1, (short)0, c11, false, false);
  }

  const float bs0 = bias[N0 + l15];
  const float bs1 = bias[N0 + 16 + l15];
  const int r0 = M0 + (hi ? 8 : 0);
  const int r1 = r0 + 16;
  const int col0 = N0 + l15, col1 = col0 + 16;
#pragma unroll
  for (int r = 0; r < 8; ++r) {
    xp[(size_t)(r0 + r) * G4 + col0] = c00[r] + bs0;
    xp[(size_t)(r0 + r) * G4 + col1] = c01[r] + bs1;
    xp[(size_t)(r1 + r) * G4 + col0] = c10[r] + bs0;
    xp[(size_t)(r1 + r) * G4 + col1] = c11[r] + bs1;
  }
}

// ---------------------------------------------------------------------------
// K4: bidirectional LSTM recurrence.
//     16 blocks x 256 thr: dir = blk/8, each of 8 WGs owns h[wg*64..wg*64+64)
//     and keeps its 256 W_hh rows resident in LDS as bf16 (~260 KB, only
//     possible with CDNA5's 320KB LDS). Staging uses async global->LDS DMA
//     (ASYNCcnt) when the toolchain exposes the gfx1250 builtin.
//     h exchanged through double-buffered global + monotonic atomic barrier.
// ---------------------------------------------------------------------------
__global__ void __launch_bounds__(256)
lstm_rec_kernel(const __bf16* __restrict__ whh16,  // [2][2048][512] bf16
                const float* __restrict__ xp,      // [2][T][2048]
                const float* __restrict__ c0,      // [2][512]
                float* __restrict__ hs,            // [2][T][512]
                float* __restrict__ hbuf,          // [2][2][512]
                unsigned int* __restrict__ bar) {
  extern __shared__ char smem[];
  __bf16* Wl = (__bf16*)smem;                          // 256 rows x WROW
  float*  hl = (float*)(smem + 256 * WROW * 2);        // 512
  float*  gl = hl + HHD;                               // 256 gate staging
  float*  cl = gl + 256;                               // 64 cell states

  const int dir = blockIdx.x >> 3;
  const int wg  = blockIdx.x & 7;
  const int tid = threadIdx.x;

  const __bf16* whh_d = whh16 + (size_t)dir * G4 * E_DIM;
  const float* xpd = xp + (size_t)dir * T_SEQ * G4;
  float* hsd = hs + (size_t)dir * T_SEQ * HHD;
  float* hb  = hbuf + (size_t)dir * 2 * HHD;

  // stage W_hh rows {q*512 + wg*64 + jj : q in 0..3, jj in 0..63} into LDS
#if HAVE_ASYNC_LDS
  // 64 b128 chunks (16B of bf16) per 512-elem row; 16384 chunks / 256 threads
  for (int c = tid; c < 256 * 64; c += 256) {
    int lr = c >> 6, cc = c & 63;
    int q = lr >> 6, jj = lr & 63;
    const __bf16* src = whh_d + (size_t)((q << 9) + (wg << 6) + jj) * HHD + (cc << 3);
    __bf16* dst = Wl + lr * WROW + (cc << 3);
    __builtin_amdgcn_global_load_async_to_lds_b128(
        (glob_v4i_p)src, (lds_v4i_p)dst, 0, 0);
  }
  __builtin_amdgcn_s_wait_asynccnt(0);
#else
  for (int i = tid; i < 256 * HHD; i += 256) {
    int lr = i >> 9, k = i & (HHD - 1);
    int q = lr >> 6, jj = lr & 63;
    Wl[lr * WROW + k] = whh_d[(size_t)((q << 9) + (wg << 6) + jj) * HHD + k];
  }
#endif
  if (tid < 64) cl[tid] = c0[dir * HHD + (wg << 6) + tid];
  __syncthreads();

  const int q  = tid >> 6;
  const int jj = tid & 63;
  const int gr = (q << 9) + (wg << 6) + jj;   // global gate row for this thread
  const uint4*  wr4 = (const uint4*)(Wl + tid * WROW);   // 16B-aligned rows
  const float4* h4  = (const float4*)hl;

  for (int s = 0; s < T_SEQ; ++s) {
    const int t = dir ? (T_SEQ - 1 - s) : s;
    const float* hread  = hb + ((s & 1) ? HHD : 0);
    float*       hwrite = hb + ((s & 1) ? 0 : HHD);

    hl[tid]       = hread[tid];
    hl[tid + 256] = hread[tid + 256];
    __syncthreads();

    float acc = xpd[(size_t)t * G4 + gr];
#pragma unroll 4
    for (int k8 = 0; k8 < HHD / 8; ++k8) {
      uint4  p  = wr4[k8];
      float4 ha = h4[2 * k8];
      float4 hbv= h4[2 * k8 + 1];
      acc = fmaf(ha.x,  bf_lo(p.x), acc);
      acc = fmaf(ha.y,  bf_hi(p.x), acc);
      acc = fmaf(ha.z,  bf_lo(p.y), acc);
      acc = fmaf(ha.w,  bf_hi(p.y), acc);
      acc = fmaf(hbv.x, bf_lo(p.z), acc);
      acc = fmaf(hbv.y, bf_hi(p.z), acc);
      acc = fmaf(hbv.z, bf_lo(p.w), acc);
      acc = fmaf(hbv.w, bf_hi(p.w), acc);
    }
    gl[tid] = acc;
    __syncthreads();

    if (tid < 64) {
      float gi = gl[tid], gf = gl[64 + tid], gg = gl[128 + tid], go = gl[192 + tid];
      float i_ = 1.0f / (1.0f + __expf(-gi));
      float f_ = 1.0f / (1.0f + __expf(-gf));
      float g_ = tanhf(gg);
      float o_ = 1.0f / (1.0f + __expf(-go));
      float c  = f_ * cl[tid] + i_ * g_;
      cl[tid] = c;
      float hv = o_ * tanhf(c);
      hwrite[(wg << 6) + tid] = hv;
      hsd[(size_t)t * HHD + (wg << 6) + tid] = hv;
    }
    __threadfence();
    __syncthreads();
    if (tid == 0) {
      atomicAdd(&bar[dir], 1u);
      const unsigned target = 8u * (unsigned)(s + 1);
      while (atomicAdd(&bar[dir], 0u) < target) __builtin_amdgcn_s_sleep(1);
      __threadfence();
    }
    __syncthreads();
  }
}

// ---------------------------------------------------------------------------
// K5: feats[t][tag] = [hs_f[t], hs_b[t]] . w_out[tag] + b_out[tag]
// ---------------------------------------------------------------------------
__global__ void feats_kernel(const float* __restrict__ hs,
                             const float* __restrict__ wout,
                             const float* __restrict__ bout,
                             float* __restrict__ feats) {
  int idx = blockIdx.x * blockDim.x + threadIdx.x;
  if (idx >= T_SEQ * NT) return;
  int t = idx / NT, tag = idx % NT;
  const float* hf = hs + (size_t)t * HHD;
  const float* hb = hs + (size_t)T_SEQ * HHD + (size_t)t * HHD;
  const float* w  = wout + (size_t)tag * (2 * HHD);
  float acc = bout[tag];
#pragma unroll 4
  for (int k = 0; k < HHD; ++k) acc += hf[k] * w[k];
#pragma unroll 4
  for (int k = 0; k < HHD; ++k) acc += hb[k] * w[HHD + k];
  feats[idx] = acc;
}

// ---------------------------------------------------------------------------
// K6: Viterbi forward (one wave32) + chunked LDS-staged backtrace
// ---------------------------------------------------------------------------
#define VCH 4096
__global__ void viterbi_kernel(const float* __restrict__ feats,
                               const float* __restrict__ trans,
                               float* __restrict__ out,       // [1+T]
                               unsigned char* __restrict__ bptr) {
  __shared__ float fv[8];
  __shared__ float tr[32];
  __shared__ unsigned char bpl[VCH * NT];
  const int tid = threadIdx.x;

  if (tid < NT * NT) tr[tid] = trans[tid];
  if (tid < NT) fv[tid] = (tid == START_TAG) ? 0.0f : NEGV;
  __syncthreads();

  for (int t = 0; t < T_SEQ; ++t) {
    float best = NEGV * 4.0f;
    int bi = 0;
    if (tid < NT) {
      best = fv[0] + tr[tid * NT + 0]; bi = 0;
#pragma unroll
      for (int i = 1; i < NT; ++i) {
        float s = fv[i] + tr[tid * NT + i];
        if (s > best) { best = s; bi = i; }
      }
      best += feats[t * NT + tid];
      bptr[t * NT + tid] = (unsigned char)bi;
    }
    __syncthreads();
    if (tid < NT) fv[tid] = best;
    __syncthreads();
  }

  __shared__ int s_tag;
  if (tid == 0) {
    float bestT = fv[0] + tr[STOP_TAG * NT + 0]; int bt = 0;
#pragma unroll
    for (int i = 1; i < NT; ++i) {
      float s = fv[i] + tr[STOP_TAG * NT + i];
      if (s > bestT) { bestT = s; bt = i; }
    }
    out[0] = bestT;
    s_tag = bt;
  }
  __syncthreads();

  for (int ch = (T_SEQ / VCH) - 1; ch >= 0; --ch) {
    for (int i = tid; i < VCH * NT; i += blockDim.x)
      bpl[i] = bptr[(size_t)ch * VCH * NT + i];
    __syncthreads();
    if (tid == 0) {
      int tag = s_tag;
      for (int t = VCH - 1; t >= 0; --t) {
        out[1 + ch * VCH + t] = (float)tag;
        tag = bpl[t * NT + tag];
      }
      s_tag = tag;
    }
    __syncthreads();
  }
}

// ---------------------------------------------------------------------------
// launcher
// ---------------------------------------------------------------------------
extern "C" void kernel_launch(void* const* d_in, const int* in_sizes, int n_in,
                              void* d_out, int out_size, void* d_ws, size_t ws_size,
                              hipStream_t stream) {
  (void)in_sizes; (void)n_in; (void)out_size; (void)ws_size;

  const int*   sentence = (const int*)  d_in[0];
  const float* embed    = (const float*)d_in[1];
  const float* w_ih_f   = (const float*)d_in[2];
  const float* w_hh_f   = (const float*)d_in[3];
  const float* b_ih_f   = (const float*)d_in[4];
  const float* b_hh_f   = (const float*)d_in[5];
  const float* w_ih_b   = (const float*)d_in[6];
  const float* w_hh_b   = (const float*)d_in[7];
  const float* b_ih_b   = (const float*)d_in[8];
  const float* b_hh_b   = (const float*)d_in[9];
  const float* w_out    = (const float*)d_in[10];
  const float* b_out    = (const float*)d_in[11];
  const float* transit  = (const float*)d_in[12];
  const float* h0       = (const float*)d_in[13];
  const float* c0       = (const float*)d_in[14];
  float* out = (float*)d_out;

  char* w = (char*)d_ws;
  __bf16* xb    = (__bf16*)w;  w += (size_t)T_SEQ * E_DIM * 2;        // 16 MB
  __bf16* wts16 = (__bf16*)w;  w += (size_t)4 * G4 * E_DIM * 2;       // 8 MB
  float* biasc  = (float*)w;   w += (size_t)2 * G4 * 4;
  float* xp     = (float*)w;   w += (size_t)2 * T_SEQ * G4 * 4;       // 256 MB
  float* hs     = (float*)w;   w += (size_t)2 * T_SEQ * HHD * 4;      // 64 MB
  float* hbuf   = (float*)w;   w += (size_t)2 * 2 * HHD * 4;
  unsigned int* bar = (unsigned int*)w; w += 256;
  unsigned char* bptr = (unsigned char*)w; w += (size_t)T_SEQ * NT;
  float* feats  = (float*)w;   w += (size_t)T_SEQ * NT * 4;

  __bf16* wih16 = wts16;                                // [2][2048][512]
  __bf16* whh16 = wts16 + (size_t)2 * G4 * E_DIM;       // [2][2048][512]

  // K0: init
  init_misc_kernel<<<16, 256, 0, stream>>>(b_ih_f, b_hh_f, b_ih_b, b_hh_b, h0,
                                           biasc, hbuf, bar);
  // K1: embedding gather -> bf16
  gather_embed_kernel<<<(T_SEQ * E_DIM) / 256, 256, 0, stream>>>(sentence, embed, xb);
  // K2: weights -> bf16
  conv_weights_kernel<<<(4 * G4 * E_DIM) / 256, 256, 0, stream>>>(w_ih_f, w_ih_b,
                                                                  w_hh_f, w_hh_b, wts16);
  // K3: input projection GEMM (both directions), WMMA bf16
  gemm_xproj_kernel<<<dim3(T_SEQ / 64, G4 / 128, 2), 256, 0, stream>>>(xb, wih16, biasc, xp);
  // K4: recurrence, 8 WGs per direction, W_hh slice resident in 320KB LDS
  const size_t rec_lds = (size_t)256 * WROW * 2 + (HHD + 256 + 64) * 4;
  lstm_rec_kernel<<<16, 256, rec_lds, stream>>>(whh16, xp, c0, hs, hbuf, bar);
  // K5: output projection
  feats_kernel<<<(T_SEQ * NT + 255) / 256, 256, 0, stream>>>(hs, w_out, b_out, feats);
  // K6: Viterbi
  viterbi_kernel<<<1, 32, 0, stream>>>(feats, transit, out, bptr);
}
